// DeepSpeedAttention_52707838657141
// MI455X (gfx1250) — matrix operand order repaired and checked by reading, not compiled, not run on hardware
//
#include <hip/hip_runtime.h>

// ---------------------------------------------------------------------------
// MI455X (gfx1250) self-attention block, bf16 WMMA everywhere.
//   B=2, S=2048, H=4096, HEADS=32, d=128.  ~687 GFLOP vs ~400MB traffic
//   => compute bound => all GEMMs on v_wmma_f32_16x16x32_bf16, flash softmax.
//   Straight-copy LDS staging uses GLOBAL_LOAD_ASYNC_TO_LDS_B128 (ASYNCcnt).
// Workspace layout (bytes), total 256 MiB:
//   [0,          32Mi)  x_bf16  [4096 x 4096]   (reused for ctx_bf16 later)
//   [32Mi,      128Mi)  Wqkv_bf16 [4096 x 12288]
//   [128Mi,     160Mi)  Wo_bf16   [4096 x 4096]
//   [160Mi,     256Mi)  q,k,v bf16, each [B*32, 2048, 128]
// ---------------------------------------------------------------------------

typedef __attribute__((ext_vector_type(8)))  float  v8f;
typedef __attribute__((ext_vector_type(8)))  __bf16 v8bf;
typedef __attribute__((ext_vector_type(16))) __bf16 v16bf;
typedef __attribute__((ext_vector_type(4)))  float  f4;
typedef __attribute__((ext_vector_type(4)))  __bf16 v4bf;

static __device__ __forceinline__ __bf16 f2bf(float f) {
  unsigned u = __float_as_uint(f);
  unsigned r = (u + 0x7FFFu + ((u >> 16) & 1u)) >> 16;
  unsigned short s = (unsigned short)r;
  return __builtin_bit_cast(__bf16, s);
}

static __device__ __forceinline__ v16bf cat8(v8bf lo, v8bf hi) {
  v16bf r;
#pragma unroll
  for (int i = 0; i < 8; ++i) { r[i] = lo[i]; r[i + 8] = hi[i]; }
  return r;
}

static __device__ __forceinline__ v8f wmma_bf16(v16bf a, v16bf b, v8f c) {
  return __builtin_amdgcn_wmma_f32_16x16x32_bf16(
      /*neg_a=*/false, a, /*neg_b=*/false, b,
      /*c_mod=*/(short)0, c, /*reuse_a=*/false, /*reuse_b=*/false);
}

// Async DMA: 16B global -> LDS, tracked on ASYNCcnt (ISA 15.18.3 op 98, GV mode).
// lds_addr = low 32 bits of generic pointer to __shared__ (the as(3) offset).
static __device__ __forceinline__ void async_b128(const void* lds_ptr,
                                                  const void* gptr) {
  unsigned lds_addr = (unsigned)(unsigned long long)lds_ptr;
  unsigned long long ga = (unsigned long long)gptr;
  asm volatile("global_load_async_to_lds_b128 %0, %1, off"
               :: "v"(lds_addr), "v"(ga)
               : "memory");
}

static __device__ __forceinline__ void wait_async0() {
#if __has_builtin(__builtin_amdgcn_s_wait_asynccnt)
  __builtin_amdgcn_s_wait_asynccnt(0);
#else
  asm volatile("s_wait_asynccnt 0x0" ::: "memory");
#endif
}

// ---------------------------------------------------------------------------
// f32 -> bf16 bulk convert (vectorized x4, grid-stride)
// ---------------------------------------------------------------------------
__global__ __launch_bounds__(256) void cvt_f32_bf16(const float* __restrict__ in,
                                                    __bf16* __restrict__ out,
                                                    size_t n4) {
  size_t i = (size_t)blockIdx.x * blockDim.x + threadIdx.x;
  size_t stride = (size_t)gridDim.x * blockDim.x;
  for (; i < n4; i += stride) {
    f4 v = ((const f4*)in)[i];
    v4bf o;
#pragma unroll
    for (int c = 0; c < 4; ++c) o[c] = f2bf(v[c]);
    ((v4bf*)out)[i] = o;
  }
}

// ---------------------------------------------------------------------------
// bf16 WMMA GEMM: C[M,N] = A[M,K] * B[K,N] + bias[N]
//   block tile 128x128, 256 threads = 8 waves (4 along M x 2 along N),
//   wave tile 32x64 = 2x4 grid of 16x16 accumulators.
// EPI==0: scatter bf16 into q/k/v per-head layout   (N = 12288)
// EPI==1: dense f32 store to out[M,N]               (N = 4096)
// ---------------------------------------------------------------------------
template <int EPI>
__global__ __launch_bounds__(256) void gemm_bf16_wmma(
    const __bf16* __restrict__ A, const __bf16* __restrict__ Bw,
    const float* __restrict__ bias, __bf16* __restrict__ outb,
    float* __restrict__ outf, int M, int N, int K) {
  // padded strides keep all ds_load_b128 16B-aligned and spread banks
  __shared__ __bf16 ldsA[128 * 40];
  __shared__ __bf16 ldsB[128 * 40];

  const int tid  = threadIdx.x;
  const int wave = tid >> 5;
  const int lane = tid & 31;
  const int ln   = lane & 15;
  const int hl   = lane >> 4;
  const int wm   = wave & 3;   // 4 waves along M
  const int wn   = wave >> 2;  // 2 waves along N

  const int m0 = blockIdx.y * 128;
  const int n0 = blockIdx.x * 128;

  const v8f vzero = {0.f, 0.f, 0.f, 0.f, 0.f, 0.f, 0.f, 0.f};
  v8f acc[2][4];
#pragma unroll
  for (int mt = 0; mt < 2; ++mt)
#pragma unroll
    for (int nt = 0; nt < 4; ++nt) acc[mt][nt] = vzero;

  for (int k0 = 0; k0 < K; k0 += 32) {
    __syncthreads();
    {  // stage A tile [128 rows][32 k] row-major via async DMA (straight copy)
      const int r = tid >> 1, half = tid & 1;
      const __bf16* src = &A[(size_t)(m0 + r) * K + k0 + half * 16];
      async_b128(&ldsA[r * 40 + half * 16], src);
      async_b128(&ldsA[r * 40 + half * 16 + 8], src + 8);
      if (k0 + 32 < K) __builtin_prefetch(src + 32, 0, 0);
    }
    {  // stage B tile transposed: ldsB[n][k], n stride 40 (VGPR permute path)
      const int kr = tid & 31, nc = tid >> 5;
      const __bf16* src = &Bw[(size_t)(k0 + kr) * N + n0 + nc * 16];
      v8bf s0 = ((const v8bf*)src)[0];
      v8bf s1 = ((const v8bf*)src)[1];
      if (k0 + 32 < K) __builtin_prefetch(src + 32 * N, 0, 0);
#pragma unroll
      for (int i = 0; i < 8; ++i) {
        ldsB[(nc * 16 + i) * 40 + kr]     = s0[i];
        ldsB[(nc * 16 + 8 + i) * 40 + kr] = s1[i];
      }
    }
    wait_async0();
    __syncthreads();

    v16bf af[2];
#pragma unroll
    for (int mt = 0; mt < 2; ++mt) {  // A frag: row=lane%16, K 8/8 split by half
      const int row = wm * 32 + mt * 16 + ln;
      const int kl  = hl * 8;
      v8bf lo = *(const v8bf*)&ldsA[row * 40 + kl];
      v8bf hi = *(const v8bf*)&ldsA[row * 40 + kl + 16];
      af[mt]  = cat8(lo, hi);
    }
    v16bf bfr[4];
#pragma unroll
    for (int nt = 0; nt < 4; ++nt) {  // B frag: n=lane%16, K 8/8 split by half
      const int col = wn * 64 + nt * 16 + ln;
      const int kl  = hl * 8;
      v8bf lo = *(const v8bf*)&ldsB[col * 40 + kl];
      v8bf hi = *(const v8bf*)&ldsB[col * 40 + kl + 16];
      bfr[nt] = cat8(lo, hi);
    }
#pragma unroll
    for (int mt = 0; mt < 2; ++mt)
#pragma unroll
      for (int nt = 0; nt < 4; ++nt)
        acc[mt][nt] = wmma_bf16(af[mt], bfr[nt], acc[mt][nt]);
  }

  // ---- epilogue: C layout row = vgpr + 8*(lane/16), col = lane%16 ----
  if constexpr (EPI == 0) {
    const int which = n0 >> 12;          // 0=q 1=k 2=v
    const int h     = (n0 & 4095) >> 7;  // head
    __bf16* base = outb + (size_t)which * (size_t)16777216;  // B*32*2048*128
#pragma unroll
    for (int mt = 0; mt < 2; ++mt)
#pragma unroll
      for (int nt = 0; nt < 4; ++nt) {
        const int n  = n0 + wn * 64 + nt * 16 + ln;
        const int dd = n & 127;
        const float bv = bias[n];
#pragma unroll
        for (int r = 0; r < 8; ++r) {
          const int m    = m0 + wm * 32 + mt * 16 + r + hl * 8;  // token
          const int bidx = m >> 11;
          const int s    = m & 2047;
          base[(((size_t)(bidx * 32 + h)) * 2048 + s) * 128 + dd] =
              f2bf(acc[mt][nt][r] + bv);
        }
      }
  } else {
#pragma unroll
    for (int mt = 0; mt < 2; ++mt)
#pragma unroll
      for (int nt = 0; nt < 4; ++nt) {
        const int n = n0 + wn * 64 + nt * 16 + ln;
        const float bv = bias[n];
#pragma unroll
        for (int r = 0; r < 8; ++r) {
          const int m = m0 + wm * 32 + mt * 16 + r + hl * 8;
          outf[(size_t)m * N + n] = acc[mt][nt][r] + bv;
        }
      }
  }
}

// ---------------------------------------------------------------------------
// Flash attention, non-causal. grid = (B*HEADS, S/128), 256 threads = 8 waves.
// Each wave owns 16 query rows; K/V streamed in 64-key blocks via LDS.
// K staged by async DMA; V transposed through VGPRs.
// scores: Q(A, resident) x K^T(B from ldsK);  ctx: P(A via ldsP) x V(B from ldsVt)
// ---------------------------------------------------------------------------
__global__ __launch_bounds__(256) void attn_flash_wmma(
    const __bf16* __restrict__ qg, const __bf16* __restrict__ kg,
    const __bf16* __restrict__ vg, __bf16* __restrict__ ctx) {
  __shared__ __bf16 ldsK[64 * 128];   // [key][d]
  __shared__ __bf16 ldsVt[128 * 72];  // [d][key], padded stride 72
  __shared__ __bf16 ldsP[8 * 16 * 64];  // per-wave P [16 rows][64 keys]

  const int tid  = threadIdx.x;
  const int wave = tid >> 5;
  const int lane = tid & 31;
  const int ln   = lane & 15;
  const int hl   = lane >> 4;

  const int bh = blockIdx.x;  // b*32 + h
  const int b  = bh >> 5;
  const int h  = bh & 31;
  const size_t base = (size_t)bh * 2048 * 128;
  const int q0 = blockIdx.y * 128;
  const float SCALE = 0.08838834764831845f;  // 1/sqrt(128)

  // resident Q fragments (A-matrix 16x32 x 4 chunks over d=128)
  v16bf qf[4];
  {
    const int qrow = q0 + wave * 16 + ln;
#pragma unroll
    for (int kc = 0; kc < 4; ++kc) {
      v8bf lo = *(const v8bf*)&qg[base + (size_t)qrow * 128 + kc * 32 + hl * 8];
      v8bf hi = *(const v8bf*)&qg[base + (size_t)qrow * 128 + kc * 32 + 16 + hl * 8];
      qf[kc] = cat8(lo, hi);
    }
  }

  const v8f vzero = {0.f, 0.f, 0.f, 0.f, 0.f, 0.f, 0.f, 0.f};
  v8f cacc[8];
#pragma unroll
  for (int dt = 0; dt < 8; ++dt) cacc[dt] = vzero;
  float m_run[8], l_run[8];
#pragma unroll
  for (int c = 0; c < 8; ++c) { m_run[c] = -INFINITY; l_run[c] = 0.f; }

  for (int kb = 0; kb < 32; ++kb) {
    __syncthreads();
    {  // stage K block [64][128] via async DMA (straight copy)
      const int row = tid >> 2, seg = tid & 3;
      const __bf16* src = &kg[base + (size_t)(kb * 64 + row) * 128 + seg * 32];
      __bf16* dst = &ldsK[row * 128 + seg * 32];
      async_b128(dst, src);
      async_b128(dst + 8, src + 8);
      async_b128(dst + 16, src + 16);
      async_b128(dst + 24, src + 24);
    }
    {  // stage V transposed: ldsVt[d][key] (VGPR permute path)
      const int key = tid >> 2, seg = tid & 3;
      const v8bf* src = (const v8bf*)&vg[base + (size_t)(kb * 64 + key) * 128 + seg * 32];
      v8bf a[4] = {src[0], src[1], src[2], src[3]};
#pragma unroll
      for (int j = 0; j < 4; ++j)
#pragma unroll
        for (int i = 0; i < 8; ++i)
          ldsVt[(seg * 32 + j * 8 + i) * 72 + key] = a[j][i];
    }
    wait_async0();
    __syncthreads();

    // ---- scores S = Q K^T * scale : 4 key-tiles x 4 d-chunks ----
    v8f sacc[4];
#pragma unroll
    for (int nt = 0; nt < 4; ++nt) sacc[nt] = vzero;
#pragma unroll
    for (int nt = 0; nt < 4; ++nt)
#pragma unroll
      for (int kc = 0; kc < 4; ++kc) {
        v8bf lo = *(const v8bf*)&ldsK[(nt * 16 + ln) * 128 + kc * 32 + hl * 8];
        v8bf hi = *(const v8bf*)&ldsK[(nt * 16 + ln) * 128 + kc * 32 + 16 + hl * 8];
        sacc[nt] = wmma_bf16(qf[kc], cat8(lo, hi), sacc[nt]);
      }
#pragma unroll
    for (int nt = 0; nt < 4; ++nt) sacc[nt] = sacc[nt] * SCALE;

    // ---- running softmax (row = vgpr + 8*hl; reduce across 16 lanes) ----
    v8f rmax = sacc[0];
#pragma unroll
    for (int nt = 1; nt < 4; ++nt)
#pragma unroll
      for (int c = 0; c < 8; ++c) rmax[c] = fmaxf(rmax[c], sacc[nt][c]);
#pragma unroll
    for (int off = 1; off < 16; off <<= 1)
#pragma unroll
      for (int c = 0; c < 8; ++c)
        rmax[c] = fmaxf(rmax[c], __shfl_xor(rmax[c], off, 32));

    float alpha[8];
#pragma unroll
    for (int c = 0; c < 8; ++c) {
      float mn = fmaxf(m_run[c], rmax[c]);
      alpha[c] = __expf(m_run[c] - mn);
      m_run[c] = mn;
    }

    float psum[8];
#pragma unroll
    for (int c = 0; c < 8; ++c) psum[c] = 0.f;
#pragma unroll
    for (int nt = 0; nt < 4; ++nt)
#pragma unroll
      for (int c = 0; c < 8; ++c) {
        float p = __expf(sacc[nt][c] - m_run[c]);
        psum[c] += p;
        ldsP[wave * 1024 + (c + hl * 8) * 64 + nt * 16 + ln] = f2bf(p);
      }
#pragma unroll
    for (int off = 1; off < 16; off <<= 1)
#pragma unroll
      for (int c = 0; c < 8; ++c) psum[c] += __shfl_xor(psum[c], off, 32);
#pragma unroll
    for (int c = 0; c < 8; ++c) l_run[c] = l_run[c] * alpha[c] + psum[c];
#pragma unroll
    for (int dt = 0; dt < 8; ++dt)
#pragma unroll
      for (int c = 0; c < 8; ++c) cacc[dt][c] *= alpha[c];

    __syncthreads();  // P store -> P fragment load

    // ---- ctx += P V : P A-frags from LDS, V B-frags from ldsVt ----
    v16bf pf[2];
#pragma unroll
    for (int pc = 0; pc < 2; ++pc) {
      v8bf lo = *(const v8bf*)&ldsP[wave * 1024 + ln * 64 + pc * 32 + hl * 8];
      v8bf hi = *(const v8bf*)&ldsP[wave * 1024 + ln * 64 + pc * 32 + 16 + hl * 8];
      pf[pc] = cat8(lo, hi);
    }
#pragma unroll
    for (int dt = 0; dt < 8; ++dt)
#pragma unroll
      for (int pc = 0; pc < 2; ++pc) {
        v8bf lo = *(const v8bf*)&ldsVt[(dt * 16 + ln) * 72 + pc * 32 + hl * 8];
        v8bf hi = *(const v8bf*)&ldsVt[(dt * 16 + ln) * 72 + pc * 32 + 16 + hl * 8];
        cacc[dt] = wmma_bf16(pf[pc], cat8(lo, hi), cacc[dt]);
      }
  }

  // ---- finalize: ctx /= l, store bf16 to [token][h*128+d] ----
  float inv[8];
#pragma unroll
  for (int c = 0; c < 8; ++c) inv[c] = 1.0f / l_run[c];
#pragma unroll
  for (int dt = 0; dt < 8; ++dt)
#pragma unroll
    for (int c = 0; c < 8; ++c) {
      const int srow  = q0 + wave * 16 + c + hl * 8;
      const int token = b * 2048 + srow;
      const int col   = h * 128 + dt * 16 + ln;
      ctx[(size_t)token * 4096 + col] = f2bf(cacc[dt][c] * inv[c]);
    }
}

// ---------------------------------------------------------------------------
extern "C" void kernel_launch(void* const* d_in, const int* in_sizes, int n_in,
                              void* d_out, int out_size, void* d_ws, size_t ws_size,
                              hipStream_t stream) {
  const float* x    = (const float*)d_in[0];
  const float* qkvw = (const float*)d_in[1];
  const float* qkvb = (const float*)d_in[2];
  const float* ow   = (const float*)d_in[3];
  const float* ob   = (const float*)d_in[4];
  float* out = (float*)d_out;

  char* ws = (char*)d_ws;
  __bf16* xb     = (__bf16*)(ws);                        // 16,777,216 elems
  __bf16* wqkvb  = (__bf16*)(ws + 33554432ull);          // 50,331,648 elems
  __bf16* wob    = (__bf16*)(ws + 134217728ull);         // 16,777,216 elems
  __bf16* qkvbuf = (__bf16*)(ws + 167772160ull);         // 3 x 16,777,216 elems
  __bf16* ctxb   = (__bf16*)(ws);                        // reuses xb region

  // 1) f32 -> bf16 converts
  cvt_f32_bf16<<<2048, 256, 0, stream>>>(x, xb, 16777216ull >> 2);
  cvt_f32_bf16<<<2048, 256, 0, stream>>>(qkvw, wqkvb, 50331648ull >> 2);
  cvt_f32_bf16<<<2048, 256, 0, stream>>>(ow, wob, 16777216ull >> 2);

  // 2) fused QKV GEMM -> per-head q/k/v (bf16)
  gemm_bf16_wmma<0><<<dim3(96, 32), 256, 0, stream>>>(
      xb, wqkvb, qkvb, qkvbuf, nullptr, 4096, 12288, 4096);

  // 3) flash attention -> ctx (bf16, [token][h*128+d])
  attn_flash_wmma<<<dim3(64, 16), 256, 0, stream>>>(
      qkvbuf, qkvbuf + 16777216, qkvbuf + 2 * 16777216, ctxb);

  // 4) output GEMM -> f32 d_out
  gemm_bf16_wmma<1><<<dim3(32, 32), 256, 0, stream>>>(
      ctxb, wob, ob, nullptr, out, 4096, 4096, 4096);
}
